// CrossAttention_678604832864
// MI455X (gfx1250) — matrix operand order repaired and checked by reading, not compiled, not run on hardware
//
#include <hip/hip_runtime.h>

// ---------------------------------------------------------------------------
// CDNA5 (gfx1250) cross-attention: bf16 WMMA GEMMs + flash-attention.
// B=8, LQ=LK=2048, QD=1024 (proj dim), KD=VD=512.
// Round 3: fix async-to-LDS builtin pointee type (v4i addrspace ptrs).
// ---------------------------------------------------------------------------

typedef __attribute__((ext_vector_type(16))) __bf16 v16bf;
typedef __attribute__((ext_vector_type(8)))  __bf16 v8bf;
typedef __attribute__((ext_vector_type(8)))  float  v8f;
typedef __attribute__((ext_vector_type(4)))  int    v4i;

#define B_   8
#define LQ_  2048
#define LK_  2048
#define D_   1024   // projected dim (QD)
#define KDIM 512    // key/value input dim

#if defined(__AMDGCN__) && __has_builtin(__builtin_amdgcn_global_load_async_to_lds_b128)
#define USE_ASYNC_LDS 1
#else
#define USE_ASYNC_LDS 0
#endif

// fp32 -> bf16, round-to-nearest-even
__device__ __forceinline__ __bf16 f2bf(float f) {
  unsigned u = __builtin_bit_cast(unsigned, f);
  unsigned r = (u + 0x7FFFu + ((u >> 16) & 1u)) >> 16;
  unsigned short s = (unsigned short)r;
  return __builtin_bit_cast(__bf16, s);
}

// 16-byte global -> LDS copy: async (ASYNCcnt-tracked) when available.
__device__ __forceinline__ void copy16_g2l(void* lds, const void* glob) {
#if USE_ASYNC_LDS
  __attribute__((address_space(1))) v4i* gp =
      (__attribute__((address_space(1))) v4i*)(unsigned long long)(uintptr_t)glob;
  __attribute__((address_space(3))) v4i* lp =
      (__attribute__((address_space(3))) v4i*)(unsigned)(uintptr_t)lds;
  __builtin_amdgcn_global_load_async_to_lds_b128(gp, lp, 0, 0);
#else
  *(uint4*)lds = *(const uint4*)glob;
#endif
}

__device__ __forceinline__ void wait_async_le1() {
#if USE_ASYNC_LDS
#if __has_builtin(__builtin_amdgcn_s_wait_asynccnt)
  __builtin_amdgcn_s_wait_asynccnt(1);
#else
  asm volatile("s_wait_asynccnt 0x1" ::: "memory");
#endif
#endif
}

__device__ __forceinline__ void wait_async_le0() {
#if USE_ASYNC_LDS
#if __has_builtin(__builtin_amdgcn_s_wait_asynccnt)
  __builtin_amdgcn_s_wait_asynccnt(0);
#else
  asm volatile("s_wait_asynccnt 0x0" ::: "memory");
#endif
#endif
}

// ---- WMMA fragment loaders (contiguous k order, same for A and B, wave32) ------------------
// A 16x32 (MxK): lane<16  -> M=lane,    elems 0..15 = K+0..15
//                lane>=16 -> M=lane-16, elems 0..15 = K+16..31
__device__ __forceinline__ v16bf load_a_bf16(const __bf16* base, int row0, int k0,
                                             int ld, int lane) {
  int m  = row0 + (lane & 15);
  int kh = (lane >> 4) * 16;
  const __bf16* p = base + (size_t)m * ld + k0 + kh;
  v8bf lo = *(const v8bf*)p;
  v8bf hi = *(const v8bf*)(p + 8);
  v16bf r;
#pragma unroll
  for (int i = 0; i < 8; ++i) { r[i] = lo[i]; r[i + 8] = hi[i]; }
  return r;
}

// Same A layout, but source is fp32 (convert in registers)
__device__ __forceinline__ v16bf load_a_f32(const float* base, int row0, int k0,
                                            int ld, int lane) {
  int m  = row0 + (lane & 15);
  int kh = (lane >> 4) * 16;
  const float* p = base + (size_t)m * ld + k0 + kh;
  v16bf r;
#pragma unroll
  for (int i = 0; i < 8; ++i) { r[i] = f2bf(p[i]); r[i + 8] = f2bf(p[i + 8]); }
  return r;
}

// B 32x16 (KxN), sourced from B^T stored row-major [N][K]:
//   lane<16 -> N=n0+lane, K = k0+0..15 ; lane>=16 -> N=n0+lane-16, K = k0+16..31
__device__ __forceinline__ v16bf load_bt_bf16(const __bf16* baseT, int n0, int k0,
                                              int ld, int lane) {
  int n = n0 + (lane & 15);
  int k = k0 + (lane >> 4) * 16;
  return *(const v16bf*)(baseT + (size_t)n * ld + k);
}

// ---------------------------------------------------------------------------
// Kernel 1: W [K][N] f32  ->  WT [N][K] bf16
// ---------------------------------------------------------------------------
__global__ void convert_wT(const float* __restrict__ W, __bf16* __restrict__ WT,
                           int K, int N) {
  int idx = blockIdx.x * 256 + threadIdx.x;
  if (idx < K * N) {
    int k = idx / N, n = idx % N;
    WT[(size_t)n * K + k] = f2bf(W[idx]);
  }
}

// ---------------------------------------------------------------------------
// Kernel 2: projection GEMM  Out = f2bf(X @ W + bias)
//   X  : [rows][K] f32      WT : [N][K] bf16      bias : [N] f32
//   transposed==0: Out[row][n]   (row-major [rows][N])
//   transposed!=0: Out[b][n][mm] (per-batch transposed, batchRows rows/batch)
// Block: 256 thr (8 waves). Block tile M=128 (16 rows/wave), N=64 (shared).
// B tile (64x32 bf16 = 4KB) staged in LDS via async copies, double-buffered.
// ---------------------------------------------------------------------------
__global__ __launch_bounds__(256) void proj_gemm(
    const float* __restrict__ X, const __bf16* __restrict__ WT,
    const float* __restrict__ bias, __bf16* __restrict__ Out,
    int K, int N, int transposed, int batchRows) {
  __shared__ __align__(32) __bf16 Bld[2][64][32];

  int lane = threadIdx.x & 31;
  int wave = threadIdx.x >> 5;
  int m0 = blockIdx.x * 128 + wave * 16;
  int n0 = blockIdx.y * 64;

  // Cooperative copier mapping: 256 threads x 16B = 4KB tile.
  int crow = threadIdx.x >> 2;        // 0..63  (WT row n0+crow)
  int ccol = (threadIdx.x & 3) * 8;   // bf16 col offset within 32-wide chunk
  const __bf16* gsrc = WT + (size_t)(n0 + crow) * K + ccol;

  copy16_g2l(&Bld[0][crow][ccol], gsrc);  // prologue: stage k0 = 0

  v8f acc[4] = {{}, {}, {}, {}};
  int buf = 0;
  for (int k0 = 0; k0 < K; k0 += 32, buf ^= 1) {
    bool more = (k0 + 32) < K;
    if (more) copy16_g2l(&Bld[buf ^ 1][crow][ccol], gsrc + k0 + 32);
    if (more) wait_async_le1(); else wait_async_le0();
    __syncthreads();  // this buffer fully staged by all waves

    v16bf a = load_a_f32(X, m0, k0, K, lane);
#pragma unroll
    for (int j = 0; j < 4; ++j) {
      const __bf16* bp = &Bld[buf][16 * j + (lane & 15)][(lane >> 4) * 16];
      v16bf b = *(const v16bf*)bp;
      acc[j] = __builtin_amdgcn_wmma_f32_16x16x32_bf16(
          false, a, false, b, (short)0, acc[j], false, false);
    }
    __syncthreads();  // done reading this buffer before it is re-staged
  }

#pragma unroll
  for (int j = 0; j < 4; ++j) {
    int n = n0 + 16 * j + (lane & 15);
    float bv = bias[n];
#pragma unroll
    for (int r = 0; r < 8; ++r) {
      int m = m0 + r + 8 * (lane >> 4);
      float v = acc[j][r] + bv;
      if (!transposed) {
        Out[(size_t)m * N + n] = f2bf(v);
      } else {
        int bb = m / batchRows, mm = m % batchRows;
        Out[(size_t)bb * N * batchRows + (size_t)n * batchRows + mm] = f2bf(v);
      }
    }
  }
}

// ---------------------------------------------------------------------------
// Kernel 3: flash attention.
//   Qp  [B][LQ][D] bf16, Kp [B][LK][D] bf16, VpT [B][D][LK] bf16 -> out f32.
// Block: 256 thr (8 waves) owns 16 query rows. Keys streamed in chunks of 128:
//   wave w computes S tile (16 x 16) for keys [kc+16w, kc+16w+16)  (32 WMMAs)
//   online softmax via LDS, then wave w accumulates O slice cols [128w,128w+128)
// ---------------------------------------------------------------------------
__global__ __launch_bounds__(256) void flash_attn(
    const __bf16* __restrict__ Qp, const __bf16* __restrict__ Kp,
    const __bf16* __restrict__ VpT, float* __restrict__ out) {
  __shared__ __align__(32) float  Sld[16][128];
  __shared__ __align__(32) __bf16 Pld[16][128];
  __shared__ float mrow[16], lrow[16], arow[16];

  int lane = threadIdx.x & 31;
  int wave = threadIdx.x >> 5;
  int batch = blockIdx.x / (LQ_ / 16);
  int qrow0 = (blockIdx.x % (LQ_ / 16)) * 16;

  const __bf16* Qb = Qp  + (size_t)batch * LQ_ * D_;
  const __bf16* Kb = Kp  + (size_t)batch * LK_ * D_;
  const __bf16* Vb = VpT + (size_t)batch * D_ * LK_;

  // Hoist the full-depth Q fragments into registers (32 x 8 VGPRs).
  v16bf qf[D_ / 32];
#pragma unroll
  for (int kk = 0; kk < D_ / 32; ++kk)
    qf[kk] = load_a_bf16(Qb, qrow0, kk * 32, D_, lane);

  v8f o[8] = {{}, {}, {}, {}, {}, {}, {}, {}};
  if (threadIdx.x < 16) { mrow[threadIdx.x] = -3.0e38f; lrow[threadIdx.x] = 0.0f; }
  __syncthreads();

  for (int kc = 0; kc < LK_; kc += 128) {
    // Warm L2/WGP$ for the next key chunk while this one computes.
    if (kc + 128 < LK_) {
      const __bf16* nk = Kb + (size_t)(kc + 128 + wave * 16 + (lane & 15)) * D_;
      __builtin_prefetch(nk, 0, 1);
    }

    // ---- S tile: Q(16 x 1024) @ K^T for this wave's 16 keys ----
    v8f s = {};
#pragma unroll
    for (int kk = 0; kk < D_ / 32; ++kk) {
      v16bf bf = load_bt_bf16(Kb, kc + wave * 16, kk * 32, D_, lane);
      s = __builtin_amdgcn_wmma_f32_16x16x32_bf16(
          false, qf[kk], false, bf, (short)0, s, false, false);
    }
#pragma unroll
    for (int r = 0; r < 8; ++r)
      Sld[r + 8 * (lane >> 4)][wave * 16 + (lane & 15)] = s[r];
    __syncthreads();

    // ---- online softmax stats (16 threads, one per query row) ----
    if (threadIdx.x < 16) {
      int t = threadIdx.x;
      float mo = mrow[t], mx = mo;
      for (int j = 0; j < 128; ++j) mx = fmaxf(mx, Sld[t][j]);
      float al = __expf(mo - mx);
      mrow[t] = mx; arow[t] = al; lrow[t] *= al;
    }
    __syncthreads();

    // ---- P = exp(S - m) as bf16; accumulate row sums (ds_add_f32) ----
    {
      int row = threadIdx.x >> 4;
      int c0  = (threadIdx.x & 15) * 8;
      float mx = mrow[row], ps = 0.0f;
#pragma unroll
      for (int i = 0; i < 8; ++i) {
        float p = __expf(Sld[row][c0 + i] - mx);
        Pld[row][c0 + i] = f2bf(p);
        ps += p;
      }
      atomicAdd(&lrow[row], ps);
    }

    // ---- rescale running O by alpha(row) ----
#pragma unroll
    for (int r = 0; r < 8; ++r) {
      float al = arow[r + 8 * (lane >> 4)];
#pragma unroll
      for (int t = 0; t < 8; ++t) o[t][r] *= al;
    }
    __syncthreads();  // Pld visible to all waves

    // ---- O += P(16x128) @ V(128 x 128-slice) ----
#pragma unroll
    for (int pk = 0; pk < 4; ++pk) {
      int mI = lane & 15, kh = (lane >> 4) * 16;
      const __bf16* pp = &Pld[mI][pk * 32 + kh];
      v8bf lo = *(const v8bf*)pp;
      v8bf hi = *(const v8bf*)(pp + 8);
      v16bf pf;
#pragma unroll
      for (int i = 0; i < 8; ++i) { pf[i] = lo[i]; pf[i + 8] = hi[i]; }
#pragma unroll
      for (int t = 0; t < 8; ++t) {
        v16bf vf = load_bt_bf16(Vb, wave * 128 + t * 16, kc + pk * 32, LK_, lane);
        o[t] = __builtin_amdgcn_wmma_f32_16x16x32_bf16(
            false, pf, false, vf, (short)0, o[t], false, false);
      }
    }
    __syncthreads();  // all waves done reading Pld/arow before next chunk
  }

  // ---- normalize by 1/l and store fp32 ----
  float* ob = out + (size_t)batch * LQ_ * D_;
#pragma unroll
  for (int r = 0; r < 8; ++r) {
    int row = r + 8 * (lane >> 4);
    float inv = 1.0f / lrow[row];
#pragma unroll
    for (int t = 0; t < 8; ++t) {
      int n = wave * 128 + t * 16 + (lane & 15);
      ob[(size_t)(qrow0 + row) * D_ + n] = o[t][r] * inv;
    }
  }
}

// ---------------------------------------------------------------------------
// Host launcher
// ---------------------------------------------------------------------------
extern "C" void kernel_launch(void* const* d_in, const int* in_sizes, int n_in,
                              void* d_out, int out_size, void* d_ws, size_t ws_size,
                              hipStream_t stream) {
  const float* query = (const float*)d_in[0];  // [8,2048,1024]
  const float* key   = (const float*)d_in[1];  // [8,2048,512]
  const float* value = (const float*)d_in[2];  // [8,2048,512]
  const float* Wq    = (const float*)d_in[3];  // [1024,1024]
  const float* bq    = (const float*)d_in[4];
  const float* Wk    = (const float*)d_in[5];  // [512,1024]
  const float* bk    = (const float*)d_in[6];
  const float* Wv    = (const float*)d_in[7];  // [512,1024]
  const float* bv    = (const float*)d_in[8];
  float* out = (float*)d_out;

  // Workspace layout (bf16 elements), ~100 MB total.
  __bf16* WqT = (__bf16*)d_ws;                        // [1024][1024]
  __bf16* WkT = WqT + (size_t)D_ * D_;                // [1024][512]
  __bf16* WvT = WkT + (size_t)D_ * KDIM;              // [1024][512]
  __bf16* Qp  = WvT + (size_t)D_ * KDIM;              // [8][2048][1024]
  __bf16* Kp  = Qp  + (size_t)B_ * LQ_ * D_;          // [8][2048][1024]
  __bf16* VpT = Kp  + (size_t)B_ * LK_ * D_;          // [8][1024][2048]

  // 1) weight transposes -> bf16
  convert_wT<<<(D_ * D_ + 255) / 256, 256, 0, stream>>>(Wq, WqT, D_, D_);
  convert_wT<<<(KDIM * D_ + 255) / 256, 256, 0, stream>>>(Wk, WkT, KDIM, D_);
  convert_wT<<<(KDIM * D_ + 255) / 256, 256, 0, stream>>>(Wv, WvT, KDIM, D_);

  // 2) projections (rows = B*L = 16384, N = 1024)
  dim3 g(16384 / 128, D_ / 64), blk(256);
  proj_gemm<<<g, blk, 0, stream>>>(query, WqT, bq, Qp, D_,   D_, 0, LQ_);
  proj_gemm<<<g, blk, 0, stream>>>(key,   WkT, bk, Kp, KDIM, D_, 0, LK_);
  proj_gemm<<<g, blk, 0, stream>>>(value, WvT, bv, VpT, KDIM, D_, 1, LK_);

  // 3) fused attention
  flash_attn<<<B_ * (LQ_ / 16), 256, 0, stream>>>(Qp, Kp, VpT, out);
}
